// PairRankGNN2_8821862826086
// MI455X (gfx1250) — matrix operand order repaired and checked by reading, not compiled, not run on hardware
//
#include <hip/hip_runtime.h>
#include <hip/hip_bf16.h>

// CDNA5 / gfx1250 wave32 WMMA types
typedef __attribute__((ext_vector_type(16))) _Float16 v16h;
typedef __attribute__((ext_vector_type(8)))  float    v8f;

#define WAVE 32

// ---------------------------------------------------------------------------
// Utility kernels
// ---------------------------------------------------------------------------
__global__ void fill_f32(float* __restrict__ p, int n, float v) {
    int i = blockIdx.x * blockDim.x + threadIdx.x;
    if (i < n) p[i] = v;
}

__global__ void relu_f32(float* __restrict__ p, int n) {
    int i = blockIdx.x * blockDim.x + threadIdx.x;
    if (i < n) p[i] = fmaxf(p[i], 0.0f);
}

// deg[dst] += 1 per edge
__global__ void deg_accum(const int* __restrict__ ei, float* __restrict__ deg, int E) {
    int e = blockIdx.x * blockDim.x + threadIdx.x;
    if (e < E) atomicAdd(&deg[ei[E + e]], 1.0f);
}

// dis = rsqrt(1 + deg)   (in place)
__global__ void deg_to_dis(float* __restrict__ d, int n) {
    int i = blockIdx.x * blockDim.x + threadIdx.x;
    if (i < n) d[i] = rsqrtf(1.0f + d[i]);
}

// ---------------------------------------------------------------------------
// GCN layer 1: xw = x @ W1 (K=9), acc = xw*dis^2 + b1  (self loop + bias)
// ---------------------------------------------------------------------------
__global__ void gcn1_init(const float* __restrict__ x, const float* __restrict__ W,
                          const float* __restrict__ b, const float* __restrict__ dis,
                          float* __restrict__ xw, float* __restrict__ acc, int N) {
    int gid = blockIdx.x * blockDim.x + threadIdx.x;
    if (gid >= N * 64) return;
    int n = gid >> 6, k = gid & 63;
    float s = 0.0f;
#pragma unroll
    for (int i = 0; i < 9; ++i) s += x[n * 9 + i] * W[i * 64 + k];
    xw[gid] = s;
    float di = dis[n];
    acc[gid] = s * di * di + b[k];
}

// GCN layer 2: xw2 = h2 @ W2 (64 -> 32), acc = xw2*dis^2 + b2
__global__ void gcn2_init(const float* __restrict__ h, const float* __restrict__ W,
                          const float* __restrict__ b, const float* __restrict__ dis,
                          float* __restrict__ xw, float* __restrict__ acc, int N) {
    int gid = blockIdx.x * blockDim.x + threadIdx.x;
    if (gid >= N * 32) return;
    int n = gid >> 5, k = gid & 31;
    float s = 0.0f;
#pragma unroll 8
    for (int i = 0; i < 64; ++i) s += h[n * 64 + i] * W[i * 32 + k];
    xw[gid] = s;
    float di = dis[n];
    acc[gid] = s * di * di + b[k];
}

// acc[dst, k..k+3] += xw[src, k..k+3] * dis[src]*dis[dst]   (float4 per thread)
template <int F>
__global__ void gcn_scatter(const float* __restrict__ xw, const float* __restrict__ dis,
                            const int* __restrict__ ei, float* __restrict__ acc, int E) {
    constexpr int Q = F / 4;
    unsigned gid = blockIdx.x * blockDim.x + threadIdx.x;
    if (gid >= (unsigned)E * Q) return;
    int e = gid / Q;
    int k = (gid & (Q - 1)) * 4;
    int s = ei[e], d = ei[E + e];
    float nrm = dis[s] * dis[d];
    const float4 v = *(const float4*)(xw + (size_t)s * F + k);
    float* base = acc + (size_t)d * F + k;
    atomicAdd(base + 0, v.x * nrm);
    atomicAdd(base + 1, v.y * nrm);
    atomicAdd(base + 2, v.z * nrm);
    atomicAdd(base + 3, v.w * nrm);
}

// ---------------------------------------------------------------------------
// EdgeConv via WMMA f32_16x16x32_f16 with DOUBLE-BUFFERED async global->LDS
// gather staging. One wave processes TPW=8 tiles of 16 edges. Raw xi/xj rows
// (f32) are DMA'd with GLOBAL_LOAD_ASYNC_TO_LDS_B128 into a ping-pong LDS
// buffer; while tile t+1's DMA is in flight (s_wait_asynccnt <= 2*CPR lets
// two CPR-op tile batches stay outstanding; ASYNCcnt completes in order) the wave
// runs WMMA on tile t. Weights are staged once per block, pre-swizzled into
// fragment-contiguous v16h order (2x ds_load_b128 per B fragment).
// out[dst] = atomic max of relu(A@W + b); out pre-zeroed, int atomicMax on
// the f32 bit pattern is exact for non-negative floats.
// ---------------------------------------------------------------------------
template <int F>
__global__ void __launch_bounds__(64)
edge_conv_wmma(const float* __restrict__ h, const float* __restrict__ W,
               const float* __restrict__ bias, const int* __restrict__ adj,
               float* __restrict__ out, int E) {
    constexpr int NW  = 2;             // waves per block
    constexpr int TPW = 8;             // tiles (16 edges each) per wave
    constexpr int F2  = 2 * F;         // inner dim of the per-edge GEMM
    constexpr int NK  = F2 / 32;       // K steps of 32
    constexpr int NT  = F / 16;        // N tiles of 16
    constexpr int CPR = F / 4;         // 16-byte chunks per row (= async ops/tile per wave, xi+xj)

    __shared__ alignas(16) float     sXi[NW][2][16 * F];  // ping-pong xi rows
    __shared__ alignas(16) float     sXj[NW][2][16 * F];  // ping-pong xj rows
    __shared__ alignas(32) _Float16  sWf[F2 * F];         // fragment-swizzled weights

    const int tid  = threadIdx.x;
    const int lane = tid & (WAVE - 1);
    const int wv   = tid >> 5;
    const int m16  = lane & 15;
    const int hi   = lane >> 4;

    // --- stage weights once per block, one v16h fragment-run per thread ----
    // run = (kk*NT + nt)*32 + ln ; element i <- W[(kk*32+(ln>>4)*8+i+(i>=8?8:0))*F + nt*16+(ln&15)]
    constexpr int RUNS = (F2 * F) / 16;
    for (int run = tid; run < RUNS; run += NW * WAVE) {
        int ln = run & 31;
        int fn = run >> 5;
        int kk = fn / NT, nt = fn - kk * NT;
        int kbase = kk * 32 + (ln >> 4) * 8;
        int n = nt * 16 + (ln & 15);
        v16h w;
#pragma unroll
        for (int i = 0; i < 16; ++i) w[i] = (_Float16)W[(kbase + i + (i >= 8 ? 8 : 0)) * F + n];
        *(v16h*)&sWf[run * 16] = w;
    }

    const int nTiles = (E + 15) / 16;
    const int t0     = (blockIdx.x * NW + wv) * TPW;

    // issue the async DMA gather for one tile into buffer `buf`
    auto issue = [&](int tile, int buf) {
        int e0 = tile * 16;
        for (int t = lane; t < 16 * CPR; t += WAVE) {
            int row = t / CPR;                 // CPR is a power of two
            int ck  = t - row * CPR;
            int e   = e0 + row;
            int s   = adj[e];                  // neighbor (xj)
            int d   = adj[E + e];              // center   (xi)
            unsigned long long gi = (unsigned long long)(h + (size_t)d * F + ck * 4);
            unsigned long long gj = (unsigned long long)(h + (size_t)s * F + ck * 4);
            unsigned li = (unsigned)(size_t)&sXi[wv][buf][t * 4];
            unsigned lj = (unsigned)(size_t)&sXj[wv][buf][t * 4];
            asm volatile("global_load_async_to_lds_b128 %0, %1, off"
                         :: "v"(li), "v"(gi) : "memory");
            asm volatile("global_load_async_to_lds_b128 %0, %1, off"
                         :: "v"(lj), "v"(gj) : "memory");
        }
    };

    if (t0 < nTiles) issue(t0, 0);     // prefetch first tile
    __syncthreads();                   // weights visible to all waves

    for (int it = 0; it < TPW; ++it) {
        const int tile = t0 + it;
        if (tile >= nTiles) break;
        const int buf = it & 1;

        const bool havenext = (it + 1 < TPW) && (tile + 1 < nTiles);
        if (havenext) {
            issue(tile + 1, buf ^ 1);
            // up to 2*CPR issues (CPR per tile for the xi/xj pair = this and the next
            // tile's DMAs) may remain outstanding -> only older tiles are sure to be in LDS
            asm volatile("s_wait_asynccnt %0" :: "i"(2 * CPR) : "memory");
        } else {
            asm volatile("s_wait_asynccnt 0x0" ::: "memory");
        }

        const int e0 = tile * 16;

        // --- build A fragments: [xi | xj-xi] in f16 -------------------------
        // 16-bit A 16x32 layout: lanes 0-15: M=lane, K=0..7 & 16..23;
        // lanes 16-31: M=lane-16, K=8..15 & 24..31 (per K-step of 32).
        v16h afr[NK];
#pragma unroll
        for (int kk = 0; kk < NK; ++kk) {
            const int part  = kk / (NK / 2);           // 0: xi half, 1: xj-xi half
            const int cbase = (kk - part * (NK / 2)) * 32;
            const float* Xi = &sXi[wv][buf][m16 * F + cbase];
            const float* Xj = &sXj[wv][buf][m16 * F + cbase];
            const int o0 = hi * 8, o1 = 16 + hi * 8;
#pragma unroll
            for (int i = 0; i < 8; ++i) {
                float a0 = Xi[o0 + i], a1 = Xi[o1 + i];
                if (part) { a0 = Xj[o0 + i] - a0; a1 = Xj[o1 + i] - a1; }
                afr[kk][i]     = (_Float16)a0;
                afr[kk][8 + i] = (_Float16)a1;
            }
        }

#pragma unroll
        for (int nt = 0; nt < NT; ++nt) {
            v8f c = {};
#pragma unroll
            for (int kk = 0; kk < NK; ++kk) {
                v16h b = *(const v16h*)&sWf[(((kk * NT) + nt) * 32 + lane) * 16];
                c = __builtin_amdgcn_wmma_f32_16x16x32_f16(
                        false, afr[kk], false, b, (short)0, c, false, false);
            }
            const float bcol = bias[nt * 16 + m16];
            // C/D layout: VGPR r -> row M = r + 8*hi, col N = m16
#pragma unroll
            for (int r = 0; r < 8; ++r) {
                int m = hi * 8 + r;
                int e = e0 + m;
                int d = adj[E + e];
                float v = fmaxf(c[r] + bcol, 0.0f);
                atomicMax((int*)(out + (size_t)d * F + nt * 16 + m16),
                          __float_as_int(v));
            }
        }
    }
}

// ---------------------------------------------------------------------------
// FC (32 -> 1) + global mean-pool accumulation
// ---------------------------------------------------------------------------
__global__ void fc_pool(const float* __restrict__ h, const float* __restrict__ fcW,
                        const float* __restrict__ fcb, const int* __restrict__ batch,
                        float* __restrict__ sums, float* __restrict__ cnts, int N) {
    int n = blockIdx.x * blockDim.x + threadIdx.x;
    if (n >= N) return;
    float s = 0.0f;
#pragma unroll 8
    for (int i = 0; i < 32; ++i) s += h[n * 32 + i] * fcW[i];
    s += fcb[0];
    int g = batch[n];
    atomicAdd(&sums[g], s);
    atomicAdd(&cnts[g], 1.0f);
}

__global__ void pool_final(const float* __restrict__ sums, const float* __restrict__ cnts,
                           float* __restrict__ out, int G) {
    int g = blockIdx.x * blockDim.x + threadIdx.x;
    if (g < G) out[g] = sums[g] / fmaxf(cnts[g], 1.0f);
}

// ---------------------------------------------------------------------------
// Launch
// ---------------------------------------------------------------------------
static inline int blks(long long n, int b) { return (int)((n + b - 1) / b); }

extern "C" void kernel_launch(void* const* d_in, const int* in_sizes, int n_in,
                              void* d_out, int out_size, void* d_ws, size_t ws_size,
                              hipStream_t stream) {
    const float* x     = (const float*)d_in[0];
    const int*   ei    = (const int*)d_in[1];
    const int*   adj   = (const int*)d_in[2];
    const int*   batch = (const int*)d_in[3];
    const float* W1  = (const float*)d_in[4];
    const float* b1  = (const float*)d_in[5];
    const float* eW1 = (const float*)d_in[6];
    const float* eb1 = (const float*)d_in[7];
    const float* W2  = (const float*)d_in[8];
    const float* b2  = (const float*)d_in[9];
    const float* eW2 = (const float*)d_in[10];
    const float* eb2 = (const float*)d_in[11];
    const float* fcW = (const float*)d_in[12];
    const float* fcb = (const float*)d_in[13];
    float* out = (float*)d_out;

    const int N = in_sizes[0] / 9;
    const int E = in_sizes[1] / 2;
    const int G = out_size;

    // workspace layout (floats)
    float* ws = (float*)d_ws;
    float* dis  = ws;                                    // N
    size_t offA = ((size_t)N + 127) & ~(size_t)127;
    float* bufA = ws + offA;                             // N*64
    float* bufB = bufA + (size_t)N * 64;                 // N*64
    float* sums = bufB + (size_t)N * 64;                 // G
    float* cnts = sums + G;                              // G

    const int B = 256;
    const int nTiles   = (E + 15) / 16;
    const int ecBlocks = (nTiles + 15) / 16;             // NW=2 waves * TPW=8 tiles

    // 1) degrees -> dis = rsqrt(1+deg)
    fill_f32<<<blks(N, B), B, 0, stream>>>(dis, N, 0.0f);
    fill_f32<<<blks(2 * G, B), B, 0, stream>>>(sums, 2 * G, 0.0f);
    deg_accum<<<blks(E, B), B, 0, stream>>>(ei, dis, E);
    deg_to_dis<<<blks(N, B), B, 0, stream>>>(dis, N);

    // 2) GCN1: xw1 -> A, acc(h1) -> B ; scatter ; relu
    gcn1_init<<<blks((long long)N * 64, B), B, 0, stream>>>(x, W1, b1, dis, bufA, bufB, N);
    gcn_scatter<64><<<blks((long long)E * 16, B), B, 0, stream>>>(bufA, dis, ei, bufB, E);
    relu_f32<<<blks((long long)N * 64, B), B, 0, stream>>>(bufB, N * 64);

    // 3) EdgeConv1 (WMMA + double-buffered async LDS gather): h1(B) -> h2(A)
    fill_f32<<<blks((long long)N * 64, B), B, 0, stream>>>(bufA, N * 64, 0.0f);
    edge_conv_wmma<64><<<ecBlocks, 64, 0, stream>>>(bufB, eW1, eb1, adj, bufA, E);

    // 4) GCN2: xw2 -> B[0..N*32), acc(h3) -> B[N*32..N*64) ; scatter ; relu
    float* xw2 = bufB;
    float* h3  = bufB + (size_t)N * 32;
    gcn2_init<<<blks((long long)N * 32, B), B, 0, stream>>>(bufA, W2, b2, dis, xw2, h3, N);
    gcn_scatter<32><<<blks((long long)E * 8, B), B, 0, stream>>>(xw2, dis, ei, h3, E);
    relu_f32<<<blks((long long)N * 32, B), B, 0, stream>>>(h3, N * 32);

    // 5) EdgeConv2 (WMMA + double-buffered async LDS gather): h3 -> h4(A)
    fill_f32<<<blks((long long)N * 32, B), B, 0, stream>>>(bufA, N * 32, 0.0f);
    edge_conv_wmma<32><<<ecBlocks, 64, 0, stream>>>(h3, eW2, eb2, adj, bufA, E);

    // 6) FC + mean pool
    fc_pool<<<blks(N, B), B, 0, stream>>>(bufA, fcW, fcb, batch, sums, cnts, N);
    pool_final<<<blks(G, B), B, 0, stream>>>(sums, cnts, out, G);
}
